// ParTAttentionBlock_31851477467191
// MI455X (gfx1250) — compile-verified, matrix operand order checked by reading, not run on hardware
//
#include <hip/hip_runtime.h>
#include <math.h>

// ---------------------------------------------------------------------------
// ParT attention block for MI455X (gfx1250, wave32, WMMA).
// bf16 operands + f32 accumulation (V_WMMA_F32_16X16X32_BF16) for all GEMMs;
// softmax/LayerNorm/GELU in f32. Flash attention with LDS-resident K/V and a
// precomputed bf16 pairwise-interaction bias.
// ---------------------------------------------------------------------------

typedef __bf16 bf16_t;
typedef __attribute__((ext_vector_type(16))) __bf16 v16bf;
typedef __attribute__((ext_vector_type(8)))  float  v8f;

#define B_   32
#define L_   512
#define D_   512
#define H_   8
#define HD_  64
#define NTOK (B_ * L_)      // 16384

// LDS strides (elements), padded for conflict-free b128 fragment reads
#define KLD  72             // K tile rows (HD_=64 + 8)
#define VLD  520            // V^T rows (L_=512 + 8)
#define PLD  40             // P restage rows (32 + 8)

// ---------------------------------------------------------------------------
// WMMA helpers
// ---------------------------------------------------------------------------
__device__ __forceinline__ v8f wmma_bf16(v16bf a, v16bf b, v8f c) {
  // 8 args: (neg_a, A, neg_b, B, c_mod, C, reuse_a, reuse_b)
  return __builtin_amdgcn_wmma_f32_16x16x32_bf16(false, a, false, b,
                                                 (short)0, c, false, false);
}

// Load one 16x32 bf16 fragment (A layout; B^T uses the same pattern) from a
// row-major tile with leading dimension `ld` (elements).
// ISA 7.12.2 16-bit A 16x32: lanes 0-15 -> row=lane, K=[0..7] and [16..23];
// lanes 16-31 -> row=lane-16, K=[8..15] and [24..31].
__device__ __forceinline__ v16bf load_frag(const bf16_t* base, int ld) {
  int lane = threadIdx.x & 31;
  int row  = lane & 15;
  int kb   = (lane >> 4) << 3;          // 0 or 8
  const bf16_t* p = base + (size_t)row * ld + kb;
  union { v16bf v; uint4 u[2]; } f;
  f.u[0] = *reinterpret_cast<const uint4*>(p);        // K +0..7
  f.u[1] = *reinterpret_cast<const uint4*>(p + 16);   // K +16..23
  return f.v;
}

// ---------------------------------------------------------------------------
// Weight convert + transpose: W (K x N, f32) -> Wt (N x K, bf16)
// ---------------------------------------------------------------------------
__global__ void cvt_transpose(const float* __restrict__ W,
                              bf16_t* __restrict__ Wt, int K, int N) {
  int idx = blockIdx.x * blockDim.x + threadIdx.x;
  if (idx >= K * N) return;
  int n = idx / K;           // consecutive idx -> consecutive k: coalesced write
  int k = idx - n * K;
  Wt[(size_t)n * K + k] = (bf16_t)W[(size_t)k * N + n];
}

// ---------------------------------------------------------------------------
// Pairwise bias precompute: inter (B,L,L,4) f32 x inter_w (4,H) ->
// biasAll (B,H,L,L) bf16. Reads inter exactly once.
// ---------------------------------------------------------------------------
__global__ void bias_precompute(const float* __restrict__ inter,
                                const float* __restrict__ inter_w,
                                bf16_t* __restrict__ biasAll) {
  size_t idx = (size_t)blockIdx.x * blockDim.x + threadIdx.x;
  const size_t total = (size_t)B_ * L_ * L_;
  if (idx >= total) return;
  const float4 iv = *reinterpret_cast<const float4*>(inter + idx * 4);
  size_t b  = idx / ((size_t)L_ * L_);
  size_t qk = idx - b * ((size_t)L_ * L_);
#pragma unroll
  for (int h = 0; h < H_; ++h) {
    float v = iv.x * inter_w[0 * H_ + h] + iv.y * inter_w[1 * H_ + h] +
              iv.z * inter_w[2 * H_ + h] + iv.w * inter_w[3 * H_ + h];
    biasAll[(b * H_ + h) * (size_t)L_ * L_ + qk] = (bf16_t)v;
  }
}

// ---------------------------------------------------------------------------
// LayerNorm over D=512, one wave per token, bf16 output.
// ---------------------------------------------------------------------------
__global__ void layernorm_bf16(const float* __restrict__ x,
                               const float* __restrict__ g,
                               const float* __restrict__ be,
                               bf16_t* __restrict__ out, int ntok) {
  int wave = (blockIdx.x * blockDim.x + threadIdx.x) >> 5;
  int lane = threadIdx.x & 31;
  if (wave >= ntok) return;
  const float* xp = x + (size_t)wave * D_;
  float vals[16];
  float s = 0.f;
#pragma unroll
  for (int i = 0; i < 16; ++i) { vals[i] = xp[lane + i * 32]; s += vals[i]; }
#pragma unroll
  for (int off = 16; off; off >>= 1) s += __shfl_xor(s, off, 32);
  float mu = s * (1.0f / D_);
  float v = 0.f;
#pragma unroll
  for (int i = 0; i < 16; ++i) { float d = vals[i] - mu; v += d * d; }
#pragma unroll
  for (int off = 16; off; off >>= 1) v += __shfl_xor(v, off, 32);
  float rstd = rsqrtf(v * (1.0f / D_) + 1e-5f);
  bf16_t* op = out + (size_t)wave * D_;
#pragma unroll
  for (int i = 0; i < 16; ++i) {
    int c = lane + i * 32;
    op[c] = (bf16_t)((vals[i] - mu) * rstd * g[c] + be[c]);
  }
}

// ---------------------------------------------------------------------------
// QKV GEMM: h1(16384x512) @ qkv_wT -> scatter into Q,K (B,H,L,64) and
// V transposed (B,H,64,L). One wave computes a 64x32 output tile (4x2 WMMA).
// ---------------------------------------------------------------------------
__global__ void __launch_bounds__(256)
gemm_qkv(const bf16_t* __restrict__ A, const bf16_t* __restrict__ Bt,
         const float* __restrict__ bias,
         bf16_t* __restrict__ Qo, bf16_t* __restrict__ Ko,
         bf16_t* __restrict__ Vto) {
  const int M = NTOK, N = 3 * D_, K = D_;
  int wave = (blockIdx.x * blockDim.x + threadIdx.x) >> 5;
  int n32 = N >> 5;
  int wm = wave / n32, wn = wave - wm * n32;
  if (wm * 64 >= M) return;
  const bf16_t* Ab = A + (size_t)wm * 64 * K;
  const bf16_t* Bb = Bt + (size_t)wn * 32 * K;
  v8f acc[4][2] = {{{}, {}}, {{}, {}}, {{}, {}}, {{}, {}}};
  for (int k = 0; k < K; k += 32) {
    __builtin_prefetch(Ab + k + 128, 0, 1);
    v16bf b0 = load_frag(Bb + k, K);
    v16bf b1 = load_frag(Bb + (size_t)16 * K + k, K);
#pragma unroll
    for (int i = 0; i < 4; ++i) {
      v16bf a = load_frag(Ab + (size_t)(i * 16) * K + k, K);
      acc[i][0] = wmma_bf16(a, b0, acc[i][0]);
      acc[i][1] = wmma_bf16(a, b1, acc[i][1]);
    }
  }
  int lane = threadIdx.x & 31;
  int hi8 = (lane >> 4) << 3;
  int nn = lane & 15;
#pragma unroll
  for (int i = 0; i < 4; ++i)
#pragma unroll
    for (int j = 0; j < 2; ++j)
#pragma unroll
      for (int r = 0; r < 8; ++r) {
        int gm = wm * 64 + i * 16 + hi8 + r;
        int gn = wn * 32 + j * 16 + nn;
        float val = acc[i][j][r] + bias[gn];
        int which = gn >> 9;            // 0=Q 1=K 2=V
        int hh = (gn >> 6) & 7;
        int dd = gn & 63;
        int bb = gm >> 9;
        int ll = gm & 511;
        size_t bh = (size_t)bb * H_ + hh;
        if (which == 0)      Qo[(bh * L_ + ll) * HD_ + dd] = (bf16_t)val;
        else if (which == 1) Ko[(bh * L_ + ll) * HD_ + dd] = (bf16_t)val;
        else                 Vto[(bh * HD_ + dd) * L_ + ll] = (bf16_t)val;
      }
}

// ---------------------------------------------------------------------------
// Flash attention, one block per (b,h). K (512x64) and V^T (64x512) are
// staged once into LDS (padded strides -> conflict-free b128 reads), then the
// 8 waves each own 4 x 16-row Q tiles. Bias comes from the precomputed bf16
// (B,H,L,L) buffer. Dynamic LDS: 512*72*2 + 64*520*2 + 8*16*40*2 bytes.
// ---------------------------------------------------------------------------
__global__ void __launch_bounds__(256)
flash_attn(const bf16_t* __restrict__ Q, const bf16_t* __restrict__ Kb,
           const bf16_t* __restrict__ Vt, const bf16_t* __restrict__ biasAll,
           bf16_t* __restrict__ O) {
  extern __shared__ char smem[];
  bf16_t* Ks = (bf16_t*)smem;                               // 512 x KLD
  bf16_t* Vs = (bf16_t*)(smem + (size_t)L_ * KLD * 2);      // 64 x VLD
  bf16_t* Ps = (bf16_t*)(smem + (size_t)L_ * KLD * 2 + (size_t)HD_ * VLD * 2);

  int bh = blockIdx.x;            // 0 .. B*H-1
  int h = bh & 7;
  int b = bh >> 3;
  int wib = threadIdx.x >> 5;
  int lane = threadIdx.x & 31;
  int hi8 = (lane >> 4) << 3;
  int nn = lane & 15;

  // ---- stage K and V^T for this (b,h) into LDS (uint4-wide) ----
  const bf16_t* Kg = Kb + (size_t)bh * L_ * HD_;
  for (int i = threadIdx.x; i < (L_ * HD_) / 8; i += 256) {   // 4096 uint4
    int row = i >> 3, q = i & 7;
    *reinterpret_cast<uint4*>(Ks + row * KLD + q * 8) =
        *reinterpret_cast<const uint4*>(Kg + row * HD_ + q * 8);
  }
  const bf16_t* Vg = Vt + (size_t)bh * HD_ * L_;
  for (int i = threadIdx.x; i < (HD_ * L_) / 8; i += 256) {   // 4096 uint4
    int row = i >> 6, q = i & 63;
    *reinterpret_cast<uint4*>(Vs + row * VLD + q * 8) =
        *reinterpret_cast<const uint4*>(Vg + row * L_ + q * 8);
  }
  __syncthreads();

  bf16_t* pb = Ps + wib * (16 * PLD);

  for (int tile = 0; tile < 4; ++tile) {
    int qt = wib * 4 + tile;                       // 0..31
    const bf16_t* Qbase = Q + ((size_t)bh * L_ + qt * 16) * HD_;
    v16bf q0 = load_frag(Qbase, HD_);              // head-dims 0..31
    v16bf q1 = load_frag(Qbase + 32, HD_);         // head-dims 32..63
    const bf16_t* bb = biasAll + ((size_t)bh * L_ + qt * 16) * L_;

    float mrow[8], lrow[8];
#pragma unroll
    for (int r = 0; r < 8; ++r) { mrow[r] = -1e30f; lrow[r] = 0.f; }
    v8f acc[4] = {{}, {}, {}, {}};

    for (int j = 0; j < L_ / 32; ++j) {
      float s[2][8];
#pragma unroll
      for (int t = 0; t < 2; ++t) {
        int kb0 = j * 32 + t * 16;
        v16bf k0 = load_frag(Ks + (size_t)kb0 * KLD, KLD);
        v16bf k1 = load_frag(Ks + (size_t)kb0 * KLD + 32, KLD);
        v8f sv = {};
        sv = wmma_bf16(q0, k0, sv);
        sv = wmma_bf16(q1, k1, sv);
#pragma unroll
        for (int r = 0; r < 8; ++r) {
          float bias = (float)bb[(size_t)(hi8 + r) * L_ + kb0 + nn];
          s[t][r] = sv[r] * 0.125f + bias;         // 1/sqrt(64)
        }
      }
      // online softmax over this 32-key chunk
#pragma unroll
      for (int r = 0; r < 8; ++r) {
        float cm = fmaxf(s[0][r], s[1][r]);
#pragma unroll
        for (int off = 8; off; off >>= 1) cm = fmaxf(cm, __shfl_xor(cm, off, 16));
        float nm = fmaxf(mrow[r], cm);
        float fac = __expf(mrow[r] - nm);
        mrow[r] = nm;
        lrow[r] *= fac;
#pragma unroll
        for (int c = 0; c < 4; ++c) acc[c][r] *= fac;
      }
      // P = exp(S - m): C layout -> row-major 16xPLD in LDS
      float ps[8];
#pragma unroll
      for (int r = 0; r < 8; ++r) ps[r] = 0.f;
#pragma unroll
      for (int t = 0; t < 2; ++t)
#pragma unroll
        for (int r = 0; r < 8; ++r) {
          float p = __expf(s[t][r] - mrow[r]);
          ps[r] += p;
          pb[(hi8 + r) * PLD + t * 16 + nn] = (bf16_t)p;
        }
#pragma unroll
      for (int r = 0; r < 8; ++r) {
        float v = ps[r];
#pragma unroll
        for (int off = 8; off; off >>= 1) v += __shfl_xor(v, off, 16);
        lrow[r] += v;
      }
      // P(16x32) @ V(32x64) from LDS
      v16bf pf = load_frag(pb, PLD);
#pragma unroll
      for (int c = 0; c < 4; ++c) {
        v16bf vf = load_frag(Vs + (size_t)(c * 16) * VLD + j * 32, VLD);
        acc[c] = wmma_bf16(pf, vf, acc[c]);
      }
    }
    // normalize and store O in (B, L, D), D = h*64 + d
#pragma unroll
    for (int c = 0; c < 4; ++c)
#pragma unroll
      for (int r = 0; r < 8; ++r) {
        float o = acc[c][r] / lrow[r];
        int row = qt * 16 + hi8 + r;
        O[((size_t)b * L_ + row) * D_ + h * HD_ + c * 16 + nn] = (bf16_t)o;
      }
  }
}

// ---------------------------------------------------------------------------
// Generic WMMA GEMM, 64x32 tile per wave. A(MxK) bf16 row-major, Bt(NxK) bf16.
// MODE 1: outF = acc + bias + resid (f32)    [out-proj -> x2, ffn2 -> d_out]
// MODE 2: outH = bf16(gelu(acc + bias))      [ffn1 -> act1]
// ---------------------------------------------------------------------------
template <int MODE>
__global__ void __launch_bounds__(256)
gemm_generic(const bf16_t* __restrict__ A, const bf16_t* __restrict__ Bt,
             const float* __restrict__ bias, const float* __restrict__ resid,
             float* __restrict__ outF, bf16_t* __restrict__ outH,
             int M, int N, int K) {
  int wave = (blockIdx.x * blockDim.x + threadIdx.x) >> 5;
  int n32 = N >> 5;
  int wm = wave / n32, wn = wave - wm * n32;
  if (wm * 64 >= M) return;
  const bf16_t* Ab = A + (size_t)wm * 64 * K;
  const bf16_t* Bb = Bt + (size_t)wn * 32 * K;
  v8f acc[4][2] = {{{}, {}}, {{}, {}}, {{}, {}}, {{}, {}}};
  for (int k = 0; k < K; k += 32) {
    __builtin_prefetch(Ab + k + 128, 0, 1);
    v16bf b0 = load_frag(Bb + k, K);
    v16bf b1 = load_frag(Bb + (size_t)16 * K + k, K);
#pragma unroll
    for (int i = 0; i < 4; ++i) {
      v16bf a = load_frag(Ab + (size_t)(i * 16) * K + k, K);
      acc[i][0] = wmma_bf16(a, b0, acc[i][0]);
      acc[i][1] = wmma_bf16(a, b1, acc[i][1]);
    }
  }
  int lane = threadIdx.x & 31;
  int hi8 = (lane >> 4) << 3;
  int nn = lane & 15;
#pragma unroll
  for (int i = 0; i < 4; ++i)
#pragma unroll
    for (int j = 0; j < 2; ++j)
#pragma unroll
      for (int r = 0; r < 8; ++r) {
        int gm = wm * 64 + i * 16 + hi8 + r;
        int gn = wn * 32 + j * 16 + nn;
        float val = acc[i][j][r] + bias[gn];
        if (MODE == 1) {
          outF[(size_t)gm * N + gn] = val + resid[(size_t)gm * N + gn];
        } else {
          float gl = 0.5f * val * (1.0f + erff(val * 0.70710678118654752f));
          outH[(size_t)gm * N + gn] = (bf16_t)gl;
        }
      }
}

// ---------------------------------------------------------------------------
// Host-side orchestration
// ---------------------------------------------------------------------------
extern "C" void kernel_launch(void* const* d_in, const int* in_sizes, int n_in,
                              void* d_out, int out_size, void* d_ws,
                              size_t ws_size, hipStream_t stream) {
  const float* x       = (const float*)d_in[0];
  const float* inter   = (const float*)d_in[1];
  const float* qkv_w   = (const float*)d_in[2];
  const float* qkv_b   = (const float*)d_in[3];
  const float* out_w   = (const float*)d_in[4];
  const float* out_b   = (const float*)d_in[5];
  const float* n1g     = (const float*)d_in[6];
  const float* n1b     = (const float*)d_in[7];
  const float* n2g     = (const float*)d_in[8];
  const float* n2b     = (const float*)d_in[9];
  const float* inter_w = (const float*)d_in[10];
  const float* f1w     = (const float*)d_in[11];
  const float* f1b     = (const float*)d_in[12];
  const float* f2w     = (const float*)d_in[13];
  const float* f2b     = (const float*)d_in[14];
  (void)in_sizes; (void)n_in; (void)out_size; (void)ws_size;

  // ---- workspace carve (phase-aliased), total ~198 MiB ----
  char* p = (char*)d_ws;
  auto take = [&](size_t bytes) { char* r = p; p += bytes; return r; };
  bf16_t* wQKVt = (bf16_t*)take((size_t)1536 * 512 * 2);
  bf16_t* wOutT = (bf16_t*)take((size_t)512 * 512 * 2);
  bf16_t* wF1T  = (bf16_t*)take((size_t)2048 * 512 * 2);
  bf16_t* wF2T  = (bf16_t*)take((size_t)512 * 2048 * 2);
  char*   hOb   = take((size_t)NTOK * D_ * 2);      // h1 (steps 2-3) / O (4-5)
  bf16_t* h1    = (bf16_t*)hOb;
  bf16_t* Ob    = (bf16_t*)hOb;
  bf16_t* Qb    = (bf16_t*)take((size_t)B_ * H_ * L_ * HD_ * 2);
  bf16_t* Kbuf  = (bf16_t*)take((size_t)B_ * H_ * L_ * HD_ * 2);
  bf16_t* Vtb   = (bf16_t*)take((size_t)B_ * H_ * HD_ * L_ * 2);
  // 128 MiB region: biasAll (steps 1-4) then x2 / h2 / act1 (steps 5-8)
  char* region = take((size_t)B_ * H_ * L_ * L_ * 2);
  bf16_t* biasAll = (bf16_t*)region;
  float*  x2   = (float*)region;                                  // 32 MiB
  bf16_t* h2   = (bf16_t*)(region + (size_t)NTOK * D_ * 4);       // 16 MiB
  bf16_t* act1 = (bf16_t*)(region + (size_t)NTOK * D_ * 4 + (size_t)NTOK * D_ * 2);

  // 1) weight convert + transpose, bias precompute
  cvt_transpose<<<(512 * 1536 + 255) / 256, 256, 0, stream>>>(qkv_w, wQKVt, 512, 1536);
  cvt_transpose<<<(512 * 512 + 255) / 256, 256, 0, stream>>>(out_w, wOutT, 512, 512);
  cvt_transpose<<<(512 * 2048 + 255) / 256, 256, 0, stream>>>(f1w, wF1T, 512, 2048);
  cvt_transpose<<<(2048 * 512 + 255) / 256, 256, 0, stream>>>(f2w, wF2T, 2048, 512);
  bias_precompute<<<(B_ * L_ * L_) / 256, 256, 0, stream>>>(inter, inter_w, biasAll);

  // 2) LN1 -> h1 (bf16)
  layernorm_bf16<<<NTOK / 8, 256, 0, stream>>>(x, n1g, n1b, h1, NTOK);

  // 3) QKV projection -> Q, K, V^T
  gemm_qkv<<<(NTOK / 64) * (1536 / 32) / 8, 256, 0, stream>>>(h1, wQKVt, qkv_b,
                                                              Qb, Kbuf, Vtb);

  // 4) flash attention (LDS-resident K/V, precomputed bias) -> O
  const size_t flash_lds = (size_t)L_ * KLD * 2 + (size_t)HD_ * VLD * 2 +
                           (size_t)8 * 16 * PLD * 2;   // 147 KiB
  flash_attn<<<B_ * H_, 256, flash_lds, stream>>>(Qb, Kbuf, Vtb, biasAll, Ob);

  // 5) out projection + residual -> x2 (f32)
  gemm_generic<1><<<(NTOK / 64) * (512 / 32) / 8, 256, 0, stream>>>(
      Ob, wOutT, out_b, x, x2, nullptr, NTOK, D_, D_);

  // 6) LN2 -> h2 (bf16)
  layernorm_bf16<<<NTOK / 8, 256, 0, stream>>>(x2, n2g, n2b, h2, NTOK);

  // 7) FFN1 + exact GELU -> act1 (bf16)
  gemm_generic<2><<<(NTOK / 64) * (2048 / 32) / 8, 256, 0, stream>>>(
      h2, wF1T, f1b, nullptr, nullptr, act1, NTOK, 4 * D_, D_);

  // 8) FFN2 + residual -> d_out (f32)
  gemm_generic<1><<<(NTOK / 64) * (512 / 32) / 8, 256, 0, stream>>>(
      act1, wF2T, f2b, x2, (float*)d_out, nullptr, NTOK, D_, 4 * D_);
}
